// TransConv_15126874817102
// MI455X (gfx1250) — matrix operand order repaired and checked by reading, not compiled, not run on hardware
//
#include <hip/hip_runtime.h>

#define N_NODES 50000
#define HID 256
#define OUTQ 1024
#define IN_CH 512
#define NP 50016              // padded node count (multiple of 32)
#define NSTEPS (NP / 32)      // 1563 k-steps of 32 for the kvs GEMM
#define KV_SLICES 8
#define KV_SPS 196            // ceil(1563/8)

typedef __attribute__((ext_vector_type(16))) __bf16 bh16;
typedef __attribute__((ext_vector_type(8)))  float  f32x8;

union BV16 { bh16 v; uint4 q[2]; };
union BV8  { uint4 q; __bf16 b[8]; };

// ---- WMMA operand loaders (CDNA5 16x16x32 bf16 layouts) ----
// A (16x32, 16-bit): lane half 0: K = [k0..k0+7] then [k0+16..k0+23];
//                    lane half 1: K = [k0+8..k0+15] then [k0+24..k0+31].
static __device__ __forceinline__ bh16 ldA16(const __bf16* __restrict__ row, int k0, int half) {
  BV16 u;
  u.q[0] = *(const uint4*)(row + k0 + half * 8);
  u.q[1] = *(const uint4*)(row + k0 + 16 + half * 8);
  return u.v;
}
// B (32x16, 16-bit): same K order as A: half 0 K=[k0..k0+7],[k0+16..k0+23]; half 1 K=[k0+8..k0+15],[k0+24..k0+31].
static __device__ __forceinline__ bh16 ldB16(const __bf16* __restrict__ row, int k0, int half) {
  BV16 u;
  u.q[0] = *(const uint4*)(row + k0 + half * 8);
  u.q[1] = *(const uint4*)(row + k0 + 16 + half * 8);
  return u.v;
}
static __device__ __forceinline__ f32x8 wmma_bf16(bh16 a, bh16 b, f32x8 c) {
  return __builtin_amdgcn_wmma_f32_16x16x32_bf16(false, a, false, b, (short)0, c, false, false);
}

// ---- prep: transpose weights [K,OUT] -> [OUT,K] and split into bf16 hi/lo ----
__global__ void wsplit_kernel(const float* __restrict__ W, int K, int OUTd,
                              __bf16* __restrict__ Thi, __bf16* __restrict__ Tlo) {
  int idx = blockIdx.x * 256 + threadIdx.x;
  if (idx >= K * OUTd) return;
  int n = idx / K, k = idx - n * K;
  float v = W[(size_t)k * OUTd + n];
  __bf16 hh = (__bf16)v;
  Thi[idx] = hh;
  Tlo[idx] = (__bf16)(v - (float)hh);
}

// ---- zero padded activation rows [N_NODES, NP) of hHi/hLo (run once) ----
__global__ void hpad_kernel(__bf16* __restrict__ hHi, __bf16* __restrict__ hLo) {
  int idx = blockIdx.x * 256 + threadIdx.x;     // 16 * 256
  if (idx >= (NP - N_NODES) * HID) return;
  size_t o = (size_t)N_NODES * HID + idx;
  hHi[o] = (__bf16)0.f;
  hLo[o] = (__bf16)0.f;
}

// ---- zero padded tail columns of Kt / Vtb (run after each projection) ----
__global__ void kvpad_kernel(__bf16* __restrict__ Kt, __bf16* __restrict__ Vtb) {
  int idx = blockIdx.x * 256 + threadIdx.x;     // 1024 * 16
  if (idx >= 1024 * 16) return;
  int row = idx >> 4, c = idx & 15;
  size_t o = (size_t)row * NP + N_NODES + c;
  Kt[o]  = (__bf16)0.f;
  Vtb[o] = (__bf16)0.f;
}

// ---- fc: x[N,512] (f32, split in-kernel) @ Wt(hi/lo)[256,512] -> hpre[N,256] ----
__global__ __launch_bounds__(256) void fc_kernel(const float* __restrict__ x,
    const __bf16* __restrict__ WtHi, const __bf16* __restrict__ WtLo,
    const float* __restrict__ bias, float* __restrict__ hpre) {
  const int tile = blockIdx.x * 8 + (threadIdx.x >> 5);   // 3125*16 tiles
  const int mt = tile >> 4, nt = tile & 15;
  const int lane = threadIdx.x & 31, lr = lane & 15, half = lane >> 4;
  const float*  xr     = x    + (size_t)(mt * 16 + lr) * IN_CH;
  const __bf16* bHiRow = WtHi + (size_t)(nt * 16 + lr) * IN_CH;
  const __bf16* bLoRow = WtLo + (size_t)(nt * 16 + lr) * IN_CH;
  f32x8 c = {};
  for (int kk = 0; kk < IN_CH / 32; ++kk) {
    const int k0 = kk * 32;
    const float4* p1 = (const float4*)(xr + k0 + half * 8);
    const float4* p2 = (const float4*)(xr + k0 + 16 + half * 8);
    float4 a0 = p1[0], a1 = p1[1], a2 = p2[0], a3 = p2[1];
    float av[16] = {a0.x,a0.y,a0.z,a0.w, a1.x,a1.y,a1.z,a1.w,
                    a2.x,a2.y,a2.z,a2.w, a3.x,a3.y,a3.z,a3.w};
    bh16 ah, al;
    #pragma unroll
    for (int e = 0; e < 16; ++e) {
      __bf16 hh = (__bf16)av[e];
      ah[e] = hh;
      al[e] = (__bf16)(av[e] - (float)hh);
    }
    bh16 bh_ = ldB16(bHiRow, k0, half);
    bh16 bl_ = ldB16(bLoRow, k0, half);
    c = wmma_bf16(ah, bh_, c);
    c = wmma_bf16(ah, bl_, c);
    c = wmma_bf16(al, bh_, c);
  }
  const int ncol = nt * 16 + lr;
  const float bb = bias[ncol];
  #pragma unroll
  for (int j = 0; j < 8; ++j)
    hpre[(size_t)(mt * 16 + half * 8 + j) * HID + ncol] = c[j] + bb;
}

// ---- LayerNorm + ReLU + bf16 hi/lo split, one wave per row ----
__global__ __launch_bounds__(256) void ln_split_kernel(const float* __restrict__ hp,
    const float* __restrict__ g, const float* __restrict__ b,
    float* __restrict__ ho, __bf16* __restrict__ hHi, __bf16* __restrict__ hLo) {
  int row = blockIdx.x * 8 + (threadIdx.x >> 5);
  int lane = threadIdx.x & 31;
  const float* pr = hp + (size_t)row * HID;
  float v[8];
  float s = 0.f;
  #pragma unroll
  for (int i = 0; i < 8; ++i) { v[i] = pr[lane + 32 * i]; s += v[i]; }
  #pragma unroll
  for (int off = 16; off; off >>= 1) s += __shfl_xor(s, off, 32);
  float mu = s * (1.0f / HID);
  float q = 0.f;
  #pragma unroll
  for (int i = 0; i < 8; ++i) { float d = v[i] - mu; q += d * d; }
  #pragma unroll
  for (int off = 16; off; off >>= 1) q += __shfl_xor(q, off, 32);
  float rstd = rsqrtf(q * (1.0f / HID) + 1e-5f);
  size_t base = (size_t)row * HID;
  #pragma unroll
  for (int i = 0; i < 8; ++i) {
    int d = lane + 32 * i;
    float y = (v[i] - mu) * rstd * g[d] + b[d];
    y = fmaxf(y, 0.f);
    ho[base + d] = y;
    __bf16 hh = (__bf16)y;
    hHi[base + d] = hh;
    hLo[base + d] = (__bf16)(y - (float)hh);
  }
}

// ---- q/k/v projections (bf16x3), 32x64 macro tile per wave.
//      Writes Qn (natural bf16), Kt (transposed bf16), Vt (transposed f32),
//      Vtb (transposed bf16), and accumulates sumsq(Q), sumsq(K), ks_sum ----
__global__ __launch_bounds__(256) void proj_kernel(
    const __bf16* __restrict__ hHi, const __bf16* __restrict__ hLo,
    const __bf16* __restrict__ wqHi, const __bf16* __restrict__ wqLo,
    const __bf16* __restrict__ wkHi, const __bf16* __restrict__ wkLo,
    const __bf16* __restrict__ wvHi, const __bf16* __restrict__ wvLo,
    const float* __restrict__ bq, const float* __restrict__ bk, const float* __restrict__ bv,
    __bf16* __restrict__ Qn, __bf16* __restrict__ Kt, float* __restrict__ Vt,
    __bf16* __restrict__ Vtb,
    float* __restrict__ sumsq_q, float* __restrict__ sumsq_k, float* __restrict__ ks_sum) {
  __shared__ float red[256];
  const int which = blockIdx.z;
  const int tid = threadIdx.x;
  const int tile = blockIdx.x * 8 + (tid >> 5);   // 1563*16 macro tiles
  const int mM = tile >> 4;                       // [0,1563): 32 rows each
  const int nM = tile & 15;                       // [0,16):   64 cols each
  const int lane = tid & 31, lr = lane & 15, half = lane >> 4;
  const __bf16 *wh, *wl; const float* bias;
  if (which == 0)      { wh = wqHi; wl = wqLo; bias = bq; }
  else if (which == 1) { wh = wkHi; wl = wkLo; bias = bk; }
  else                 { wh = wvHi; wl = wvLo; bias = bv; }
  const __bf16* aHi0 = hHi + (size_t)(mM * 32 + lr) * HID;
  const __bf16* aLo0 = hLo + (size_t)(mM * 32 + lr) * HID;
  const __bf16* aHi1 = hHi + (size_t)(mM * 32 + 16 + lr) * HID;
  const __bf16* aLo1 = hLo + (size_t)(mM * 32 + 16 + lr) * HID;
  const __bf16* bHiR[4]; const __bf16* bLoR[4];
  #pragma unroll
  for (int id = 0; id < 4; ++id) {
    bHiR[id] = wh + (size_t)(nM * 64 + id * 16 + lr) * HID;
    bLoR[id] = wl + (size_t)(nM * 64 + id * 16 + lr) * HID;
  }
  f32x8 cc[2][4] = {};
  for (int kk = 0; kk < HID / 32; ++kk) {
    const int k0 = kk * 32;
    bh16 ah0 = ldA16(aHi0, k0, half);
    bh16 al0 = ldA16(aLo0, k0, half);
    bh16 ah1 = ldA16(aHi1, k0, half);
    bh16 al1 = ldA16(aLo1, k0, half);
    #pragma unroll
    for (int id = 0; id < 4; ++id) {
      bh16 bh_ = ldB16(bHiR[id], k0, half);
      bh16 bl_ = ldB16(bLoR[id], k0, half);
      cc[0][id] = wmma_bf16(ah0, bh_, cc[0][id]);
      cc[0][id] = wmma_bf16(ah0, bl_, cc[0][id]);
      cc[0][id] = wmma_bf16(al0, bh_, cc[0][id]);
      cc[1][id] = wmma_bf16(ah1, bh_, cc[1][id]);
      cc[1][id] = wmma_bf16(ah1, bl_, cc[1][id]);
      cc[1][id] = wmma_bf16(al1, bh_, cc[1][id]);
    }
  }
  // add bias
  #pragma unroll
  for (int id = 0; id < 4; ++id) {
    const float bb = bias[nM * 64 + id * 16 + lr];
    #pragma unroll
    for (int im = 0; im < 2; ++im)
      #pragma unroll
      for (int j = 0; j < 8; ++j) cc[im][id][j] += bb;
  }

  if (which == 0) {                       // Q: natural layout bf16 + sumsq
    #pragma unroll
    for (int im = 0; im < 2; ++im)
      #pragma unroll
      for (int id = 0; id < 4; ++id) {
        const int ncol = nM * 64 + id * 16 + lr;
        #pragma unroll
        for (int j = 0; j < 8; ++j)
          Qn[(size_t)(mM * 32 + im * 16 + half * 8 + j) * OUTQ + ncol] = (__bf16)cc[im][id][j];
      }
    float sq = 0.f;
    #pragma unroll
    for (int im = 0; im < 2; ++im)
      #pragma unroll
      for (int id = 0; id < 4; ++id)
        #pragma unroll
        for (int j = 0; j < 8; ++j) sq += cc[im][id][j] * cc[im][id][j];
    red[tid] = sq; __syncthreads();
    for (int s = 128; s; s >>= 1) { if (tid < s) red[tid] += red[tid + s]; __syncthreads(); }
    if (tid == 0) atomicAdd(sumsq_q, red[0]);
  } else if (which == 1) {                // K: transposed bf16 + stats
    #pragma unroll
    for (int im = 0; im < 2; ++im)
      #pragma unroll
      for (int id = 0; id < 4; ++id) {
        const int ncol = nM * 64 + id * 16 + lr;
        BV8 pk;
        #pragma unroll
        for (int j = 0; j < 8; ++j) pk.b[j] = (__bf16)cc[im][id][j];
        *(uint4*)(Kt + (size_t)ncol * NP + mM * 32 + im * 16 + half * 8) = pk.q;
      }
    #pragma unroll
    for (int id = 0; id < 4; ++id) {
      float s1 = 0.f;
      #pragma unroll
      for (int im = 0; im < 2; ++im)
        #pragma unroll
        for (int j = 0; j < 8; ++j) s1 += cc[im][id][j];
      atomicAdd(&ks_sum[nM * 64 + id * 16 + lr], s1);
    }
    float sq = 0.f;
    #pragma unroll
    for (int im = 0; im < 2; ++im)
      #pragma unroll
      for (int id = 0; id < 4; ++id)
        #pragma unroll
        for (int j = 0; j < 8; ++j) sq += cc[im][id][j] * cc[im][id][j];
    red[tid] = sq; __syncthreads();
    for (int s = 128; s; s >>= 1) { if (tid < s) red[tid] += red[tid + s]; __syncthreads(); }
    if (tid == 0) atomicAdd(sumsq_k, red[0]);
  } else {                                // V: transposed f32 + transposed bf16
    #pragma unroll
    for (int im = 0; im < 2; ++im)
      #pragma unroll
      for (int id = 0; id < 4; ++id) {
        const int ncol = nM * 64 + id * 16 + lr;
        const size_t mo = (size_t)(mM * 32 + im * 16 + half * 8);
        float4 v0 = {cc[im][id][0], cc[im][id][1], cc[im][id][2], cc[im][id][3]};
        float4 v1 = {cc[im][id][4], cc[im][id][5], cc[im][id][6], cc[im][id][7]};
        float* vp = Vt + (size_t)ncol * NP + mo;
        *(float4*)vp       = v0;
        *(float4*)(vp + 4) = v1;
        BV8 pk;
        #pragma unroll
        for (int j = 0; j < 8; ++j) pk.b[j] = (__bf16)cc[im][id][j];
        *(uint4*)(Vtb + (size_t)ncol * NP + mo) = pk.q;
      }
  }
}

// ---- s = 1/(||Q|| * ||K||) ----
__global__ void scal_kernel(float* __restrict__ stats) {
  if (threadIdx.x == 0) stats[2] = rsqrtf(stats[0] * stats[1]);
}

// ---- kvs_un[h][d][m] = sum_l K[l,hm] * V[l,hd]
//      32(m) x 64(d) macro tile per wave, split-K over 8 slices, f32 atomics ----
__global__ __launch_bounds__(256) void kvs_kernel(const __bf16* __restrict__ Kt,
    const __bf16* __restrict__ Vtb, float* __restrict__ kvsAcc) {
  const int t = blockIdx.x * 8 + (threadIdx.x >> 5);   // 1024 waves
  const int slice = t >> 7;                            // 8 slices
  const int tf = t & 127;
  const int h = tf >> 5, mt = (tf >> 2) & 7, dt = tf & 3;
  const int lane = threadIdx.x & 31, lr = lane & 15, half = lane >> 4;
  const __bf16* aR0 = Kt + (size_t)(h * 256 + mt * 32 + lr) * NP;
  const __bf16* aR1 = Kt + (size_t)(h * 256 + mt * 32 + 16 + lr) * NP;
  const __bf16* bR[4];
  #pragma unroll
  for (int id = 0; id < 4; ++id)
    bR[id] = Vtb + (size_t)(h * 256 + dt * 64 + id * 16 + lr) * NP;
  int s0 = slice * KV_SPS;
  int s1 = s0 + KV_SPS; if (s1 > NSTEPS) s1 = NSTEPS;
  f32x8 cc[2][4] = {};
  for (int st = s0; st < s1; ++st) {
    const int l0 = st * 32;
    bh16 a0 = ldA16(aR0, l0, half);
    bh16 a1 = ldA16(aR1, l0, half);
    #pragma unroll
    for (int id = 0; id < 4; ++id) {
      bh16 b = ldB16(bR[id], l0, half);
      cc[0][id] = wmma_bf16(a0, b, cc[0][id]);
      cc[1][id] = wmma_bf16(a1, b, cc[1][id]);
    }
  }
  #pragma unroll
  for (int im = 0; im < 2; ++im)
    #pragma unroll
    for (int id = 0; id < 4; ++id) {
      float* dst = kvsAcc + (size_t)(h * 256 + dt * 64 + id * 16 + lr) * 256
                 + mt * 32 + im * 16 + half * 8;
      #pragma unroll
      for (int j = 0; j < 8; ++j) atomicAdd(dst + j, cc[im][id][j]);
    }
}

// ---- scale kvs by s -> bf16; t = ks_sum * s ----
__global__ void kscale_kernel(const float* __restrict__ kvsAcc, const float* __restrict__ stats,
                              const float* __restrict__ ks_sum, __bf16* __restrict__ kvsb,
                              float* __restrict__ tarr) {
  int idx = blockIdx.x * 256 + threadIdx.x;   // 262144 exactly
  float sv = stats[2];
  kvsb[idx] = (__bf16)(kvsAcc[idx] * sv);
  if (idx < 1024) tarr[idx] = ks_sum[idx] * sv;
}

// ---- fused attention pass: Q@kvs_s (WMMA, A reused across 8 tiles) + N*V,
//      /denom, head-mean, residual, LayerNorm, ReLU, bf16 re-split ----
__global__ __launch_bounds__(256) void attn_kernel(
    const __bf16* __restrict__ Qn, const float* __restrict__ Vt,
    const __bf16* __restrict__ kvsb, const float* __restrict__ tarr,
    const float* __restrict__ hin, const float* __restrict__ g, const float* __restrict__ bt,
    float* __restrict__ hout, __bf16* __restrict__ hHi, __bf16* __restrict__ hLo) {
  __shared__ float outAcc[16 * 256];
  __shared__ float tbuf[1024];
  __shared__ float dinv[64];
  __shared__ float redS[256];
  __shared__ float muR[16], rsR[16];
  const int tid = threadIdx.x;
  const int m0 = blockIdx.x * 16;
  for (int i = tid; i < 1024; i += 256) tbuf[i] = tarr[i];
  for (int i = tid; i < 4096; i += 256) outAcc[i] = 0.f;
  __syncthreads();
  // denominators: denom[n,h] = Q[n,h,:]·t[h,:] + N
  if (tid < 64) {
    int r = tid & 15, h = tid >> 4;
    const __bf16* q = Qn + (size_t)(m0 + r) * OUTQ + h * 256;
    const float* tp = tbuf + h * 256;
    float dot = 0.f;
    for (int mm = 0; mm < 256; mm += 8) {
      BV8 u; u.q = *(const uint4*)(q + mm);
      #pragma unroll
      for (int e = 0; e < 8; ++e) dot += (float)u.b[e] * tp[mm + e];
    }
    dinv[tid] = 1.0f / (dot + (float)N_NODES);
  }
  __syncthreads();
  {
    const int wave = tid >> 5, lane = tid & 31, lr = lane & 15, half = lane >> 4;
    const int h = wave >> 1, dtB = (wave & 1) * 8;   // 8 d-tiles, one head per wave
    const __bf16* aRow  = Qn   + (size_t)(m0 + lr) * OUTQ + h * 256;
    const __bf16* bRow0 = kvsb + (size_t)(h * 256 + dtB * 16 + lr) * 256;
    f32x8 cc[8] = {};
    #pragma unroll
    for (int kk = 0; kk < 8; ++kk) {
      bh16 a = ldA16(aRow, kk * 32, half);
      #pragma unroll
      for (int i = 0; i < 8; ++i)
        cc[i] = wmma_bf16(a, ldB16(bRow0 + (size_t)i * 16 * 256, kk * 32, half), cc[i]);
    }
    #pragma unroll
    for (int i = 0; i < 8; ++i) {
      const int dt = dtB + i;
      const float* vp = Vt + (size_t)(h * 256 + dt * 16 + lr) * NP + m0 + half * 8;
      float4 v0 = *(const float4*)vp;
      float4 v1 = *(const float4*)(vp + 4);
      float vv[8] = {v0.x, v0.y, v0.z, v0.w, v1.x, v1.y, v1.z, v1.w};
      #pragma unroll
      for (int j = 0; j < 8; ++j) {
        const int r2 = half * 8 + j;
        float val = (cc[i][j] + (float)N_NODES * vv[j]) * dinv[h * 16 + r2] * 0.25f;
        atomicAdd(&outAcc[r2 * 256 + dt * 16 + lr], val);
      }
    }
  }
  __syncthreads();
  // epilogue: residual + LayerNorm + ReLU + split
  const int r = tid >> 4, sub = tid & 15;
  const size_t node = (size_t)(m0 + r);
  const float* pv = hin + node * HID;
  float myres[16]; float s1 = 0.f;
  #pragma unroll
  for (int i = 0; i < 16; ++i) {
    const int d = sub * 16 + i;
    float res = (outAcc[r * 256 + d] + pv[d]) * 0.5f;
    myres[i] = res; s1 += res;
  }
  redS[r * 16 + sub] = s1; __syncthreads();
  if (sub == 0) { float mu = 0.f; for (int k2 = 0; k2 < 16; ++k2) mu += redS[r * 16 + k2]; muR[r] = mu * (1.0f / HID); }
  __syncthreads();
  const float mu = muR[r];
  float q2 = 0.f;
  #pragma unroll
  for (int i = 0; i < 16; ++i) { float d0 = myres[i] - mu; q2 += d0 * d0; }
  redS[r * 16 + sub] = q2; __syncthreads();
  if (sub == 0) { float v = 0.f; for (int k2 = 0; k2 < 16; ++k2) v += redS[r * 16 + k2]; rsR[r] = rsqrtf(v * (1.0f / HID) + 1e-5f); }
  __syncthreads();
  const float rs = rsR[r];
  #pragma unroll
  for (int i = 0; i < 16; ++i) {
    const int d = sub * 16 + i;
    float y = (myres[i] - mu) * rs * g[d] + bt[d];
    y = fmaxf(y, 0.f);
    hout[node * HID + d] = y;
    __bf16 hh = (__bf16)y;
    hHi[node * HID + d] = hh;
    hLo[node * HID + d] = (__bf16)(y - (float)hh);
  }
}

extern "C" void kernel_launch(void* const* d_in, const int* in_sizes, int n_in,
                              void* d_out, int out_size, void* d_ws, size_t ws_size,
                              hipStream_t stream) {
  const float* x    = (const float*)d_in[0];
  const float* fc_w = (const float*)d_in[1];
  const float* fc_b = (const float*)d_in[2];
  const float* ln_g[3] = {(const float*)d_in[3], (const float*)d_in[11], (const float*)d_in[19]};
  const float* ln_b[3] = {(const float*)d_in[4], (const float*)d_in[12], (const float*)d_in[20]};
  const float* wq[2] = {(const float*)d_in[5],  (const float*)d_in[13]};
  const float* bq[2] = {(const float*)d_in[6],  (const float*)d_in[14]};
  const float* wk[2] = {(const float*)d_in[7],  (const float*)d_in[15]};
  const float* bk[2] = {(const float*)d_in[8],  (const float*)d_in[16]};
  const float* wv[2] = {(const float*)d_in[9],  (const float*)d_in[17]};
  const float* bv[2] = {(const float*)d_in[10], (const float*)d_in[18]};

  char* ws = (char*)d_ws;
  size_t off = 0;
  auto alloc = [&](size_t bytes) -> char* {
    char* p = ws + off;
    off = (off + bytes + 255) & ~(size_t)255;
    return p;
  };
  float*  hA    = (float*) alloc((size_t)N_NODES * HID * 4);
  float*  hB    = (float*) alloc((size_t)N_NODES * HID * 4);
  __bf16* hHi   = (__bf16*)alloc((size_t)NP * HID * 2);     // padded rows
  __bf16* hLo   = (__bf16*)alloc((size_t)NP * HID * 2);
  __bf16* Qn    = (__bf16*)alloc((size_t)NP * OUTQ * 2);    // padded rows
  __bf16* Kt    = (__bf16*)alloc((size_t)OUTQ * NP * 2);
  float*  Vt    = (float*) alloc((size_t)OUTQ * NP * 4);
  __bf16* Vtb   = (__bf16*)alloc((size_t)OUTQ * NP * 2);
  __bf16* fcwHi = (__bf16*)alloc((size_t)IN_CH * HID * 2);
  __bf16* fcwLo = (__bf16*)alloc((size_t)IN_CH * HID * 2);
  __bf16* wHi[2][3];
  __bf16* wLo[2][3];
  for (int L = 0; L < 2; ++L)
    for (int m = 0; m < 3; ++m) {
      wHi[L][m] = (__bf16*)alloc((size_t)HID * OUTQ * 2);
      wLo[L][m] = (__bf16*)alloc((size_t)HID * OUTQ * 2);
    }
  float*  kvsAcc = (float*) alloc((size_t)262144 * 4);
  __bf16* kvsb   = (__bf16*)alloc((size_t)262144 * 2);
  float*  ks_sum = (float*) alloc(4096);
  float*  tarr   = (float*) alloc(4096);
  float*  stats  = (float*) alloc(256);   // [0]=sumsq_q [1]=sumsq_k [2]=s

  // ---- prep: weight transpose + hi/lo split, pad rows (tiny, rerun every launch) ----
  wsplit_kernel<<<(IN_CH * HID + 255) / 256, 256, 0, stream>>>(fc_w, IN_CH, HID, fcwHi, fcwLo);
  const float* wmat[2][3] = {{wq[0], wk[0], wv[0]}, {wq[1], wk[1], wv[1]}};
  for (int L = 0; L < 2; ++L)
    for (int m = 0; m < 3; ++m)
      wsplit_kernel<<<(HID * OUTQ + 255) / 256, 256, 0, stream>>>(wmat[L][m], HID, OUTQ, wHi[L][m], wLo[L][m]);
  hpad_kernel<<<16, 256, 0, stream>>>(hHi, hLo);

  // ---- input FFN: fc + LN + ReLU ----
  fc_kernel<<<6250, 256, 0, stream>>>(x, fcwHi, fcwLo, fc_b, hB);
  ln_split_kernel<<<6250, 256, 0, stream>>>(hB, ln_g[0], ln_b[0], hA, hHi, hLo);

  // ---- two transformer-conv layers ----
  for (int L = 0; L < 2; ++L) {
    hipMemsetAsync(kvsAcc, 0, (size_t)262144 * 4, stream);
    hipMemsetAsync(ks_sum, 0, 4096, stream);
    hipMemsetAsync(stats, 0, 16, stream);
    proj_kernel<<<dim3(3126, 1, 3), 256, 0, stream>>>(
        hHi, hLo,
        wHi[L][0], wLo[L][0], wHi[L][1], wLo[L][1], wHi[L][2], wLo[L][2],
        bq[L], bk[L], bv[L],
        Qn, Kt, Vt, Vtb, stats + 0, stats + 1, ks_sum);
    kvpad_kernel<<<64, 256, 0, stream>>>(Kt, Vtb);   // re-zero pad cols written by pad rows
    scal_kernel<<<1, 32, 0, stream>>>(stats);
    kvs_kernel<<<dim3(128, 1), 256, 0, stream>>>(Kt, Vtb, kvsAcc);
    kscale_kernel<<<1024, 256, 0, stream>>>(kvsAcc, stats, ks_sum, kvsb, tarr);
    const float* hin = (L == 0) ? hA : hB;
    float* hout = (L == 0) ? hB : (float*)d_out;
    attn_kernel<<<3125, 256, 0, stream>>>(Qn, Vt, kvsb, tarr, hin,
                                          ln_g[L + 1], ln_b[L + 1], hout, hHi, hLo);
  }
}
